// MultiHeadAttention_69466801046036
// MI455X (gfx1250) — matrix operand order repaired and checked by reading, not compiled, not run on hardware
//
#include <hip/hip_runtime.h>
#include <hip/hip_bf16.h>

#define NUM_HEADS 16
#define HEAD_SIZE 64
#define EMBED     1024
#define BATCH     4
#define SEQ       2048

typedef __attribute__((ext_vector_type(16))) _Float16 v16h;
typedef __attribute__((ext_vector_type(8)))  float    v8f;

// ---------------------------------------------------------------------------
// WMMA helpers (V_WMMA_F32_16X16X32_F16, fp32 accumulate)
// ---------------------------------------------------------------------------
__device__ __forceinline__ v8f wmma_f16(v16h a, v16h b, v8f c) {
  return __builtin_amdgcn_wmma_f32_16x16x32_f16(
      /*neg_a=*/false, a, /*neg_b=*/false, b,
      /*c_mod=*/(short)0, c, /*reuse_a=*/false, /*reuse_b=*/false);
}

// A fragment: row-major MxK source, 16x32 tile at (row0, k0).
// Layout: lanes 0-15 -> M=lane, halves[0..15]=K0..15;
//         lanes 16-31 -> K16..31.
__device__ __forceinline__ v16h frag_a(const _Float16* base, int stride,
                                       int row0, int k0) {
  const int lane = threadIdx.x & 31;
  const int row  = row0 + (lane & 15);
  const int kb   = k0 + ((lane >> 4) << 4);
  const _Float16* p = base + (size_t)row * stride + kb;
  v16h f;
#pragma unroll
  for (int j = 0; j < 8; ++j) f[j] = p[j];
#pragma unroll
  for (int j = 0; j < 8; ++j) f[8 + j] = p[8 + j];
  return f;
}

// B fragment from a K-contiguous store: lds[n*stride + k] holds B[K=k][N=n].
// Layout (contiguous k, as in frag_a): lanes 0-15 -> N=lane, halves[0..15]=K0..15; lanes 16-31 -> K16..31.
__device__ __forceinline__ v16h frag_b(const _Float16* base, int stride,
                                       int col0, int k0) {
  const int lane = threadIdx.x & 31;
  const int col  = col0 + (lane & 15);
  const int kb   = k0 + ((lane >> 4) << 4);
  const _Float16* p = base + (size_t)col * stride + kb;
  v16h f;
#pragma unroll
  for (int j = 0; j < 16; ++j) f[j] = p[j];
  return f;
}

// ---------------------------------------------------------------------------
// Kernel 1: QKV projection.  q/k/v[b,h,t,e] = sum_d x[b,t,d] * W{q,k,v}[h,d,e]
// Block: 256 thr (8 waves) -> 128(T) x 64(hs) tile. grid(16, H, B*3).
// ---------------------------------------------------------------------------
#define XSTR 36   // 32 + 4 pad halves: 72B rows (8B aligned, conflict-free)
#define WSTR 36   // W^T tile: [e][kk], K-contiguous

__global__ __launch_bounds__(256) void qkv_kernel(
    const float* __restrict__ x,  const float* __restrict__ Wq,
    const float* __restrict__ Wk, const float* __restrict__ Wv,
    _Float16* __restrict__ qh, _Float16* __restrict__ kh,
    _Float16* __restrict__ vh) {
  __shared__ _Float16 lx[128 * XSTR];
  __shared__ _Float16 lw[64 * WSTR];

  const int t0   = blockIdx.x * 128;
  const int h    = blockIdx.y;
  const int b    = blockIdx.z / 3;
  const int mat  = blockIdx.z % 3;
  const float* W   = (mat == 0) ? Wq : (mat == 1) ? Wk : Wv;
  _Float16*   out  = (mat == 0) ? qh : (mat == 1) ? kh : vh;

  const int tid = threadIdx.x, wave = tid >> 5, lane = tid & 31;
  const int strip = wave * 16;

  v8f acc[4] = {};

  for (int k0 = 0; k0 < EMBED; k0 += 32) {
    __syncthreads();
    // x tile: 128 rows x 32 k, fp32 -> f16
#pragma unroll
    for (int it = 0; it < 16; ++it) {
      int idx = tid + it * 256;             // 4096 elements
      int r = idx >> 5, c = idx & 31;
      lx[r * XSTR + c] =
          (_Float16)x[((size_t)b * SEQ + t0 + r) * EMBED + k0 + c];
    }
    // W tile transposed: lw[e][kk] = W[h, k0+kk, e]
#pragma unroll
    for (int it = 0; it < 8; ++it) {
      int idx = tid + it * 256;             // 2048 elements
      int kk = idx >> 6, e = idx & 63;
      lw[e * WSTR + kk] =
          (_Float16)W[((size_t)h * EMBED + k0 + kk) * HEAD_SIZE + e];
    }
    __syncthreads();

    v16h a = frag_a(lx, XSTR, strip, 0);
#pragma unroll
    for (int nt = 0; nt < 4; ++nt) {
      v16h bf = frag_b(lw, WSTR, nt * 16, 0);
      acc[nt] = wmma_f16(a, bf, acc[nt]);
    }
  }

  // store f16 [B,H,T,hs]
  const int n = lane & 15, mb = (lane >> 4) << 3;
  size_t base =
      (((size_t)b * NUM_HEADS + h) * SEQ + t0 + strip + mb) * HEAD_SIZE;
#pragma unroll
  for (int nt = 0; nt < 4; ++nt)
#pragma unroll
    for (int i = 0; i < 8; ++i)
      out[base + (size_t)i * HEAD_SIZE + nt * 16 + n] = (_Float16)acc[nt][i];
}

// ---------------------------------------------------------------------------
// Kernel 2: causal flash attention over f16 q/k/v, fp32 online softmax.
// Block: 128 thr (4 waves); each wave = one 16-row query strip of a 64-row
// tile.  grid(T/64, H, B).
// ---------------------------------------------------------------------------
#define KVSTR 72  // 64 + 8 pad halves: 144B rows (8B aligned, conflict-free)
#define PSTR  72

__global__ __launch_bounds__(128) void attn_kernel(
    const _Float16* __restrict__ qh, const _Float16* __restrict__ kh,
    const _Float16* __restrict__ vh, _Float16* __restrict__ atth) {
  __shared__ _Float16 lk [64 * KVSTR];        // K tile  [tk][e]
  __shared__ _Float16 lvt[64 * KVSTR];        // V tile transposed [e][tk]
  __shared__ _Float16 lp [4][16 * PSTR];      // per-wave P strip [m][tk]

  const int qt = blockIdx.x, h = blockIdx.y, b = blockIdx.z;
  const int tid = threadIdx.x, wave = tid >> 5, lane = tid & 31;
  const int strip = wave * 16;

  const size_t headBase = ((size_t)b * NUM_HEADS + h) * SEQ * HEAD_SIZE;
  const _Float16* qp = qh + headBase + (size_t)(qt * 64) * HEAD_SIZE;

  // Q fragments live in registers for the whole block (K=hs=64 -> 2 chunks)
  v16h qf[2];
  qf[0] = frag_a(qp, HEAD_SIZE, strip, 0);
  qf[1] = frag_a(qp, HEAD_SIZE, strip, 32);

  v8f   o[4] = {};
  float m_i[8], l_i[8];
#pragma unroll
  for (int i = 0; i < 8; ++i) { m_i[i] = -1e30f; l_i[i] = 0.f; }

  const int mb   = (lane >> 4) << 3;
  const int qrow = qt * 64 + strip + mb;      // +i gives this lane's rows

  for (int j = 0; j <= qt; ++j) {
    const _Float16* kp = kh + headBase + (size_t)(j * 64) * HEAD_SIZE;
    const _Float16* vp = vh + headBase + (size_t)(j * 64) * HEAD_SIZE;
    __syncthreads();
#pragma unroll
    for (int it = 0; it < 8; ++it) {
      int idx = (tid + it * 128) * 4;         // 4096 halves
      int r = idx >> 6, c = idx & 63;
      *(uint2*)&lk[r * KVSTR + c] = *(const uint2*)&kp[(size_t)r * 64 + c];
      uint2 vv = *(const uint2*)&vp[(size_t)r * 64 + c];
      const _Float16* v4 = (const _Float16*)&vv;
#pragma unroll
      for (int q2 = 0; q2 < 4; ++q2) lvt[(c + q2) * KVSTR + r] = v4[q2];
    }
    __syncthreads();

    if (j < qt) {  // prefetch next K/V tile (global_prefetch_b8)
      __builtin_prefetch(kp + 64 * HEAD_SIZE + (size_t)tid * 32, 0, 1);
      __builtin_prefetch(vp + 64 * HEAD_SIZE + (size_t)tid * 32, 0, 1);
    }

    // S = Q * K^T   (B element (K=e, N=tk) = lk[tk*KVSTR + e])
    v8f s[4] = {};
#pragma unroll
    for (int kc = 0; kc < 2; ++kc)
#pragma unroll
      for (int nt = 0; nt < 4; ++nt) {
        v16h bf = frag_b(lk, KVSTR, nt * 16, kc * 32);
        s[nt] = wmma_f16(qf[kc], bf, s[nt]);
      }

    // scale + causal mask + row max
    const float scale = 0.125f;               // hs^-0.5
    float rmax[8];
#pragma unroll
    for (int i = 0; i < 8; ++i) {
      float mx = -1e30f;
#pragma unroll
      for (int nt = 0; nt < 4; ++nt) {
        float v = s[nt][i] * scale;
        if (j == qt) {
          int kg = j * 64 + nt * 16 + (lane & 15);
          if (kg > qrow + i) v = -1e30f;
        }
        s[nt][i] = v;
        mx = fmaxf(mx, v);
      }
      rmax[i] = mx;
    }
#pragma unroll
    for (int d = 1; d < 16; d <<= 1)
#pragma unroll
      for (int i = 0; i < 8; ++i)
        rmax[i] = fmaxf(rmax[i], __shfl_xor(rmax[i], d, 32));

    float alpha[8], rsum[8];
#pragma unroll
    for (int i = 0; i < 8; ++i) {
      float mn = fmaxf(m_i[i], rmax[i]);
      alpha[i] = __expf(m_i[i] - mn);
      m_i[i]   = mn;
      rsum[i]  = 0.f;
    }

    // P = exp(S - m) -> per-wave LDS strip (A operand of P*V)
#pragma unroll
    for (int nt = 0; nt < 4; ++nt)
#pragma unroll
      for (int i = 0; i < 8; ++i) {
        float p = __expf(s[nt][i] - m_i[i]);
        rsum[i] += p;
        lp[wave][(mb + i) * PSTR + nt * 16 + (lane & 15)] = (_Float16)p;
      }
#pragma unroll
    for (int d = 1; d < 16; d <<= 1)
#pragma unroll
      for (int i = 0; i < 8; ++i) rsum[i] += __shfl_xor(rsum[i], d, 32);
#pragma unroll
    for (int i = 0; i < 8; ++i) l_i[i] = l_i[i] * alpha[i] + rsum[i];

    // rescale O, then O += P * V  (B element (K=tk, N=e) = lvt[e*KVSTR+tk])
#pragma unroll
    for (int nt = 0; nt < 4; ++nt)
#pragma unroll
      for (int i = 0; i < 8; ++i) o[nt][i] *= alpha[i];
#pragma unroll
    for (int kc = 0; kc < 2; ++kc) {
      v16h pa = frag_a(lp[wave], PSTR, 0, kc * 32);
#pragma unroll
      for (int nt = 0; nt < 4; ++nt) {
        v16h bf = frag_b(lvt, KVSTR, nt * 16, kc * 32);
        o[nt] = wmma_f16(pa, bf, o[nt]);
      }
    }
  }

  // epilogue: normalize, write concat-head layout [B,T,D] as f16
  const int n = lane & 15;
#pragma unroll
  for (int i = 0; i < 8; ++i) {
    float inv = 1.0f / l_i[i];
    int t = qt * 64 + strip + mb + i;
    size_t rowb = ((size_t)b * SEQ + t) * EMBED + h * HEAD_SIZE;
#pragma unroll
    for (int nt = 0; nt < 4; ++nt)
      atth[rowb + nt * 16 + n] = (_Float16)(o[nt][i] * inv);
  }
}

// ---------------------------------------------------------------------------
// Kernel 3: out = att @ Wo^T + bo (fp32 out).  Block 256 thr -> 128x64 tile.
// grid(B*T/128, D/64).
// ---------------------------------------------------------------------------
#define ASTR 36
#define BSTR 36

__global__ __launch_bounds__(256) void proj_kernel(
    const _Float16* __restrict__ atth, const float* __restrict__ Wo,
    const float* __restrict__ bo, float* __restrict__ out) {
  __shared__ _Float16 la[128 * ASTR];
  __shared__ _Float16 lb[64 * BSTR];   // lb[i][kk] = Wo[n0+i, k0+kk]

  const int r0 = blockIdx.x * 128;     // over B*T
  const int n0 = blockIdx.y * 64;      // over D
  const int tid = threadIdx.x, wave = tid >> 5, lane = tid & 31;
  const int strip = wave * 16;

  v8f acc[4] = {};

  for (int k0 = 0; k0 < EMBED; k0 += 32) {
    __syncthreads();
#pragma unroll
    for (int it = 0; it < 4; ++it) {
      int idx = (tid + it * 256) * 4;  // 4096 halves
      int r = idx >> 5, c = idx & 31;
      *(uint2*)&la[r * ASTR + c] =
          *(const uint2*)&atth[(size_t)(r0 + r) * EMBED + k0 + c];
    }
#pragma unroll
    for (int it = 0; it < 8; ++it) {
      int idx = tid + it * 256;        // 2048 elements
      int i = idx >> 5, kk = idx & 31;
      lb[i * BSTR + kk] = (_Float16)Wo[(size_t)(n0 + i) * EMBED + k0 + kk];
    }
    __syncthreads();

    v16h a = frag_a(la, ASTR, strip, 0);
#pragma unroll
    for (int nt = 0; nt < 4; ++nt) {
      v16h bf = frag_b(lb, BSTR, nt * 16, 0);
      acc[nt] = wmma_f16(a, bf, acc[nt]);
    }
  }

  const int n = lane & 15, mb = (lane >> 4) << 3;
#pragma unroll
  for (int nt = 0; nt < 4; ++nt) {
    float bias = bo[n0 + nt * 16 + n];
#pragma unroll
    for (int i = 0; i < 8; ++i)
      out[(size_t)(r0 + strip + mb + i) * EMBED + n0 + nt * 16 + n] =
          acc[nt][i] + bias;
  }
}

// ---------------------------------------------------------------------------
extern "C" void kernel_launch(void* const* d_in, const int* in_sizes, int n_in,
                              void* d_out, int out_size, void* d_ws,
                              size_t ws_size, hipStream_t stream) {
  const float* x  = (const float*)d_in[0];
  const float* Wq = (const float*)d_in[1];
  const float* Wk = (const float*)d_in[2];
  const float* Wv = (const float*)d_in[3];
  const float* Wo = (const float*)d_in[4];
  const float* bo = (const float*)d_in[5];
  float* out = (float*)d_out;

  const size_t perBuf = (size_t)BATCH * NUM_HEADS * SEQ * HEAD_SIZE;  // 8M halves
  _Float16* qh   = (_Float16*)d_ws;
  _Float16* kh   = qh + perBuf;
  _Float16* vh   = kh + perBuf;
  _Float16* atth = vh + perBuf;       // [B,T,D] f16, 8M halves

  qkv_kernel<<<dim3(SEQ / 128, NUM_HEADS, BATCH * 3), 256, 0, stream>>>(
      x, Wq, Wk, Wv, qh, kh, vh);
  attn_kernel<<<dim3(SEQ / 64, NUM_HEADS, BATCH), 128, 0, stream>>>(
      qh, kh, vh, atth);
  proj_kernel<<<dim3(BATCH * SEQ / 128, EMBED / 64), 256, 0, stream>>>(
      atth, Wo, bo, out);
}
